// GraphAttentionLayer_20667382629035
// MI455X (gfx1250) — matrix operand order repaired and checked by reading, not compiled, not run on hardware
//
#include <hip/hip_runtime.h>
#include <hip/hip_bf16.h>

typedef __attribute__((ext_vector_type(2)))  float    v2f;
typedef __attribute__((ext_vector_type(4)))  float    v4f;
typedef __attribute__((ext_vector_type(8)))  float    v8f;
typedef __attribute__((ext_vector_type(4)))  int      v4i;
typedef __attribute__((ext_vector_type(8)))  _Float16 v8h;
typedef __attribute__((ext_vector_type(16))) _Float16 v16h;

#define N_NODES 8192
#define F_IN    256
#define F_OUT   64
#define F_E     8
#define ALPHA   0.2f
#define NEG_INF (-3.0e38f)

// ---------------------------------------------------------------------------
// Kernel 1: Wh = h @ W (fp32-exact via v_wmma_f32_16x16x4_f32)
// Also writes WhT (f16, transposed 64 x 8192) for fast B-tile loads later.
// Grid: 512 blocks x 128 threads (4 waves). Wave w -> N-tile [16w,16w+16).
// ---------------------------------------------------------------------------
__global__ __launch_bounds__(128) void k_wh(const float* __restrict__ h,
                                            const float* __restrict__ W,
                                            float* __restrict__ Wh,
                                            _Float16* __restrict__ WhT) {
  const int wave  = threadIdx.x >> 5;
  const int lane  = threadIdx.x & 31;
  const int rb    = blockIdx.x * 16;
  const int row   = rb + (lane & 15);            // A: lane holds row M = lane%16
  const int khalf = (lane >> 4) * 2;             // A: lanes>=16 hold K=2,3
  const int ncol  = wave * 16 + (lane & 15);     // B/C: lane holds col N = lane%16

  v8f c = {};
  for (int k = 0; k < F_IN; k += 4) {
    v2f a = *(const v2f*)(h + (size_t)row * F_IN + k + khalf);
    v2f b;
    b.x = W[(size_t)(k + khalf + 0) * F_OUT + ncol];
    b.y = W[(size_t)(k + khalf + 1) * F_OUT + ncol];
    c = __builtin_amdgcn_wmma_f32_16x16x4_f32(false, a, false, b,
                                              (short)0, c, false, false);
  }
  const int roff = (lane >> 4) * 8;              // C: lanes>=16 hold M = r+8
#pragma unroll
  for (int r = 0; r < 8; ++r) {
    const int orow = rb + r + roff;
    Wh[(size_t)orow * F_OUT + ncol] = c[r];
    WhT[(size_t)ncol * N_NODES + orow] = (_Float16)c[r];
  }
}

// ---------------------------------------------------------------------------
// Kernel 2: src[i] = Wh[i].a1 + edge[i].(We@a3);  dst[i] = Wh[i].a2
// ---------------------------------------------------------------------------
__global__ __launch_bounds__(256) void k_srcdst(const float* __restrict__ Wh,
                                                const float* __restrict__ edge,
                                                const float* __restrict__ a,
                                                const float* __restrict__ We,
                                                float* __restrict__ srcv,
                                                float* __restrict__ dstv) {
  const int i = blockIdx.x * blockDim.x + threadIdx.x;
  if (i >= N_NODES) return;
  const float* a1 = a;
  const float* a2 = a + F_OUT;
  const float* a3 = a + 2 * F_OUT;
  float s1 = 0.f, s2 = 0.f;
#pragma unroll 8
  for (int n = 0; n < F_OUT; ++n) {
    const float w = Wh[(size_t)i * F_OUT + n];
    s1 += w * a1[n];
    s2 += w * a2[n];
  }
  float se = 0.f;
  for (int f = 0; f < F_E; ++f) {
    float s3 = 0.f;
#pragma unroll 8
    for (int n = 0; n < F_OUT; ++n) s3 += We[f * F_OUT + n] * a3[n];
    se += edge[i * F_E + f] * s3;
  }
  srcv[i] = s1 + se;
  dstv[i] = s2;
}

// ---------------------------------------------------------------------------
// Kernel 3: fused masked softmax + attention@Wh (flash-style, split-K x 4 waves)
// Block = 16 rows. Each wave processes its own 32-column slab per iteration,
// keeping a private online-softmax state (m, l) and a 16x64 fp32 accumulator,
// and accumulates via v_wmma_f32_16x16x32_f16. Final 4-way merge via LDS.
// ---------------------------------------------------------------------------
__device__ inline v8f wmma_btile(v16h A, const _Float16* __restrict__ bp, v8f acc) {
  const v8h blo = *(const v8h*)(bp);
  const v8h bhi = *(const v8h*)(bp + 16);
  v16h B;
#pragma unroll
  for (int i = 0; i < 8; ++i) { B[i] = blo[i]; B[i + 8] = bhi[i]; }
  return __builtin_amdgcn_wmma_f32_16x16x32_f16(false, A, false, B,
                                                (short)0, acc, false, false);
}

__global__ __launch_bounds__(128) void k_attn(const int* __restrict__ adj,
                                              const float* __restrict__ srcv,
                                              const float* __restrict__ dstv,
                                              const _Float16* __restrict__ WhT,
                                              float* __restrict__ out) {
  __shared__ float lds_m[4][16];
  __shared__ float lds_l[4][16];
  __shared__ float lds_acc[4][16][64];

  const int wave = threadIdx.x >> 5;
  const int lane = threadIdx.x & 31;
  const int rb   = blockIdx.x * 16;
  const int mrow = lane & 15;               // row this lane tracks (A layout)
  const int klo  = (lane < 16) ? 0 : 8;     // A-matrix K groups for this lane
  const int khi  = klo + 16;
  const int kb   = (lane < 16) ? 0 : 8;     // B-matrix K group for this lane
  const int ncol = lane & 15;               // B/C column within a 16-wide tile

  float m = NEG_INF;
  float l = 0.0f;
  v8f acc0 = {}, acc1 = {}, acc2 = {}, acc3 = {};

  const float s_i = srcv[rb + mrow];
  const int* __restrict__ adjrow = adj + (size_t)(rb + mrow) * N_NODES;

  for (int it = 0; it < N_NODES / 128; ++it) {
    const int c0 = it * 128 + wave * 32;    // this wave's 32-col slab

    // adjacency: each element read exactly once across the wave (b128 loads)
    const v4i q0 = *(const v4i*)(adjrow + c0 + klo);
    const v4i q1 = *(const v4i*)(adjrow + c0 + klo + 4);
    const v4i q2 = *(const v4i*)(adjrow + c0 + khi);
    const v4i q3 = *(const v4i*)(adjrow + c0 + khi + 4);
    const v4f d0 = *(const v4f*)(dstv + c0 + klo);
    const v4f d1 = *(const v4f*)(dstv + c0 + klo + 4);
    const v4f d2 = *(const v4f*)(dstv + c0 + khi);
    const v4f d3 = *(const v4f*)(dstv + c0 + khi + 4);

    int   msk[16];
    float e[16];
#pragma unroll
    for (int i = 0; i < 4; ++i) {
      msk[i]      = q0[i];        msk[i + 4]  = q1[i];
      msk[i + 8]  = q2[i];        msk[i + 12] = q3[i];
      e[i]        = s_i + d0[i];  e[i + 4]    = s_i + d1[i];
      e[i + 8]    = s_i + d2[i];  e[i + 12]   = s_i + d3[i];
    }

    float tmax = NEG_INF;
#pragma unroll
    for (int i = 0; i < 16; ++i) {
      float x = e[i];
      x = (x >= 0.0f) ? x : ALPHA * x;            // leaky_relu
      x = (msk[i] > 0) ? x : NEG_INF;             // adjacency mask
      e[i] = x;
      tmax = fmaxf(tmax, x);
    }
    // rows live in lane L and L+16 -> combine halves
    tmax = fmaxf(tmax, __shfl_xor(tmax, 16, 32));
    const float mnew  = fmaxf(m, tmax);
    const float scale = __expf(m - mnew);

    float psum = 0.0f;
    v16h A;                                        // p already in A lane layout
#pragma unroll
    for (int i = 0; i < 16; ++i) {
      const float p = (msk[i] > 0) ? __expf(e[i] - mnew) : 0.0f;
      psum += p;
      A[i] = (_Float16)p;
    }
    psum += __shfl_xor(psum, 16, 32);
    l = l * scale + psum;
    m = mnew;

    // rescale accumulator rows: C layout row = r + 8*(lane>=16)
    const int half8 = (lane >> 4) << 3;
#pragma unroll
    for (int r = 0; r < 8; ++r) {
      const float sr = __shfl(scale, r + half8, 32);
      acc0[r] *= sr; acc1[r] *= sr; acc2[r] *= sr; acc3[r] *= sr;
    }

    // P(16x32) @ WhT-slab(32x64) via 4 WMMAs (fp32 accumulate)
    const _Float16* __restrict__ bbase = WhT + c0 + kb;
    acc0 = wmma_btile(A, bbase + (size_t)(ncol     ) * N_NODES, acc0);
    acc1 = wmma_btile(A, bbase + (size_t)(ncol + 16) * N_NODES, acc1);
    acc2 = wmma_btile(A, bbase + (size_t)(ncol + 32) * N_NODES, acc2);
    acc3 = wmma_btile(A, bbase + (size_t)(ncol + 48) * N_NODES, acc3);
  }

  // ----- split-K merge across the 4 waves -----
  if (lane < 16) { lds_m[wave][lane] = m; lds_l[wave][lane] = l; }
  const int roff = (lane >> 4) * 8;
#pragma unroll
  for (int r = 0; r < 8; ++r) {
    const int rr = r + roff;
    lds_acc[wave][rr][ncol     ] = acc0[r];
    lds_acc[wave][rr][ncol + 16] = acc1[r];
    lds_acc[wave][rr][ncol + 32] = acc2[r];
    lds_acc[wave][rr][ncol + 48] = acc3[r];
  }
  __syncthreads();

  for (int e0 = threadIdx.x; e0 < 16 * 64; e0 += 128) {
    const int row = e0 >> 6, col = e0 & 63;
    const float M = fmaxf(fmaxf(lds_m[0][row], lds_m[1][row]),
                          fmaxf(lds_m[2][row], lds_m[3][row]));
    float L = 0.f, V = 0.f;
#pragma unroll
    for (int w = 0; w < 4; ++w) {
      const float f = __expf(lds_m[w][row] - M);
      L += lds_l[w][row] * f;
      V += lds_acc[w][row][col] * f;
    }
    const float hp = V / L;
    out[(size_t)(rb + row) * F_OUT + col] = (hp > 0.0f) ? hp : expm1f(hp); // elu
  }
}

// ---------------------------------------------------------------------------
extern "C" void kernel_launch(void* const* d_in, const int* in_sizes, int n_in,
                              void* d_out, int out_size, void* d_ws, size_t ws_size,
                              hipStream_t stream) {
  const float* h    = (const float*)d_in[0];
  const float* edge = (const float*)d_in[1];
  const int*   adj  = (const int*)  d_in[2];
  const float* W    = (const float*)d_in[3];
  const float* a    = (const float*)d_in[4];
  const float* We   = (const float*)d_in[5];
  float* out = (float*)d_out;

  char* ws = (char*)d_ws;
  float*    Wh   = (float*)ws;                                        // 2 MB
  _Float16* WhT  = (_Float16*)(ws + (size_t)N_NODES * F_OUT * 4);     // 1 MB
  float*    srcv = (float*)(ws + (size_t)3 * 1024 * 1024);            // 32 KB
  float*    dstv = srcv + N_NODES;                                    // 32 KB

  k_wh    <<<N_NODES / 16, 128, 0, stream>>>(h, W, Wh, WhT);
  k_srcdst<<<N_NODES / 256, 256, 0, stream>>>(Wh, edge, a, We, srcv, dstv);
  k_attn  <<<N_NODES / 16, 128, 0, stream>>>(adj, srcv, dstv, WhT, out);
}
